// Proofreader_60095182405725
// MI455X (gfx1250) — matrix operand order repaired and checked by reading, not compiled, not run on hardware
//
#include <hip/hip_runtime.h>
#include <hip/hip_bf16.h>

typedef __attribute__((ext_vector_type(16))) _Float16 v16h;
typedef __attribute__((ext_vector_type(8)))  _Float16 v8h;
typedef __attribute__((ext_vector_type(8)))  float    v8f;

#define CLASSN 3815
#define CP     3840   // class dim padded to multiple of 128
#define VEC    1024
#define HID    512
#define BATCH  4096
#define TSTEPS 5      // only t=0..4 (fwd) / t=4..8 (bwd) are needed
#define SEQT   9

// ---------------------------------------------------------------------------
// Generic WMMA GEMM:  out = act( A[M,K](f16, optional row-gather) * W[N,K]^T
//                               + bias[n] + residual[m,n] ) * mul[m,n]
// Block: 256 thr = 8 waves (4 M x 2 N), wave tile 32x64 -> block tile 128x128.
// Requires M % 128 == 0, NP % 128 == 0, K % 32 == 0.
// Inner loop: 12 x global_load_b128 per 8 x v_wmma_f32_16x16x32_f16.
// ---------------------------------------------------------------------------
__global__ __launch_bounds__(256)
void gemm_f16_wmma(const _Float16* __restrict__ A, int lda,
                   const int* __restrict__ gidx,
                   const _Float16* __restrict__ W, int ldb,
                   int N, int K,
                   const float* __restrict__ bias,
                   const float* __restrict__ resid, int ldr,
                   const float* __restrict__ mul, int ldm,
                   int act,                      // 0 none, 1 tanh, 2 sigmoid
                   float* __restrict__ outF, int ldoF,
                   _Float16* __restrict__ outH, int ldoH)
{
    const int lane = threadIdx.x & 31;
    const int wave = threadIdx.x >> 5;
    const int wm = wave >> 1;            // 0..3
    const int wn = wave & 1;             // 0..1
    const int tile_m = blockIdx.x * 128 + wm * 32;
    const int tile_n = blockIdx.y * 128 + wn * 64;
    const int lrow = lane & 15;
    const int aksel = (lane < 16) ? 0 : 8;    // A frag K select per half-wave
    const int bksel = (lane < 16) ? 0 : 8;    // B frag K select per half-wave

    int arow[2];
#pragma unroll
    for (int i = 0; i < 2; ++i) {
        int r = tile_m + 16 * i + lrow;
        arow[i] = gidx ? gidx[r] : r;
    }
    int bcol[4];
#pragma unroll
    for (int j = 0; j < 4; ++j) bcol[j] = tile_n + 16 * j + lrow;

    v8f acc[2][4] = {};

    for (int k = 0; k < K; k += 32) {
        v16h a[2], b[4];
#pragma unroll
        for (int i = 0; i < 2; ++i) {
            const _Float16* p = A + (size_t)arow[i] * lda + k + aksel;
            v8h lo = *(const v8h*)p;
            v8h hi = *(const v8h*)(p + 16);
            a[i] = __builtin_shufflevector(lo, hi, 0,1,2,3,4,5,6,7,8,9,10,11,12,13,14,15);
        }
#pragma unroll
        for (int j = 0; j < 4; ++j) {
            const _Float16* p = W + (size_t)bcol[j] * ldb + k + bksel;
            v8h lo = *(const v8h*)p;
            v8h hi = *(const v8h*)(p + 16);
            b[j] = __builtin_shufflevector(lo, hi, 0,1,2,3,4,5,6,7,8,9,10,11,12,13,14,15);
        }
#pragma unroll
        for (int i = 0; i < 2; ++i)
#pragma unroll
            for (int j = 0; j < 4; ++j)
                acc[i][j] = __builtin_amdgcn_wmma_f32_16x16x32_f16(
                    false, a[i], false, b[j], (short)0, acc[i][j], false, false);
    }

    const int madd = (lane < 16) ? 0 : 8;
#pragma unroll
    for (int i = 0; i < 2; ++i) {
#pragma unroll
        for (int j = 0; j < 4; ++j) {
            const int gn = tile_n + 16 * j + lrow;
#pragma unroll
            for (int e = 0; e < 8; ++e) {
                const int gm = tile_m + 16 * i + e + madd;
                float v = acc[i][j][e];
                if (gn < N) {
                    if (bias)  v += bias[gn];
                    if (resid) v += resid[(size_t)gm * ldr + gn];
                    if (act == 1)      v = tanhf(v);
                    else if (act == 2) v = 1.f / (1.f + expf(-v));
                    if (mul)   v *= mul[(size_t)gm * ldm + gn];
                    if (outF) outF[(size_t)gm * ldoF + gn] = v;
                    if (outH) outH[(size_t)gm * ldoH + gn] = (_Float16)v;
                } else if (outH) {
                    outH[(size_t)gm * ldoH + gn] = (_Float16)0.f;   // zero K-padding
                }
            }
        }
    }
}

// Row-normalize mean_vector -> f16 copies (padded rows zeroed).
__global__ __launch_bounds__(256)
void normalize_kernel(const float* __restrict__ mv,
                      _Float16* __restrict__ mvn16, _Float16* __restrict__ mv16)
{
    __shared__ float red[256];
    const int row = blockIdx.x;            // 0..CP-1
    const int t = threadIdx.x;
    if (row >= CLASSN) {
        for (int c = t; c < VEC; c += 256) {
            mvn16[(size_t)row * VEC + c] = (_Float16)0.f;
            mv16 [(size_t)row * VEC + c] = (_Float16)0.f;
        }
        return;
    }
    const float* src = mv + (size_t)row * VEC;
    float s = 0.f;
    for (int c = t; c < VEC; c += 256) { float x = src[c]; s += x * x; }
    red[t] = s; __syncthreads();
    for (int off = 128; off > 0; off >>= 1) {
        if (t < off) red[t] += red[t + off];
        __syncthreads();
    }
    const float scale = 1.f / fmaxf(sqrtf(red[0]), 1e-8f);
    for (int c = t; c < VEC; c += 256) {
        float x = src[c];
        mvn16[(size_t)row * VEC + c] = (_Float16)(x * scale);
        mv16 [(size_t)row * VEC + c] = (_Float16)x;
    }
}

// f32 -> f16 with zero padding: dst[RP x CS] from src[R x C]
__global__ void cvt_pad_kernel(const float* __restrict__ src, int R, int C,
                               _Float16* __restrict__ dst, int RP, int CS)
{
    size_t i = (size_t)blockIdx.x * blockDim.x + threadIdx.x;
    size_t total = (size_t)RP * CS;
    if (i >= total) return;
    int r = (int)(i / CS), c = (int)(i % CS);
    dst[i] = (r < R && c < C) ? (_Float16)src[(size_t)r * C + c] : (_Float16)0.f;
}

// Gather indices: idx_f[t*B+b] = x[b,t] (t=0..4); idx_b[s*B+b] = x[b,4+s]
__global__ void build_idx_kernel(const int* __restrict__ x,
                                 int* __restrict__ idx_f, int* __restrict__ idx_b)
{
    int i = blockIdx.x * blockDim.x + threadIdx.x;
    if (i >= TSTEPS * BATCH) return;
    int t = i >> 12, b = i & (BATCH - 1);
    idx_f[i] = x[b * SEQT + t];
    idx_b[i] = x[b * SEQT + 4 + t];
}

__global__ void bias_kernel(const float* a, const float* b,
                            const float* c, const float* d,
                            float* bf, float* bb)
{
    int i = blockIdx.x * blockDim.x + threadIdx.x;
    if (i < HID) { bf[i] = a[i] + b[i]; bb[i] = c[i] + d[i]; }
}

__global__ void zero_h_kernel(_Float16* p, size_t n)
{
    size_t i = (size_t)blockIdx.x * blockDim.x + threadIdx.x;
    if (i < n) p[i] = (_Float16)0.f;
}

// out4 = concat(h_f[4], h_b[4]) in f16
__global__ void pack_kernel(const _Float16* __restrict__ hf,
                            const _Float16* __restrict__ hb,
                            _Float16* __restrict__ out4)
{
    int i = blockIdx.x * blockDim.x + threadIdx.x;
    if (i >= BATCH * 2 * HID) return;
    int b = i >> 10, j = i & 1023;
    out4[i] = (j < HID) ? hf[b * HID + j] : hb[b * HID + (j - HID)];
}

extern "C" void kernel_launch(void* const* d_in, const int* in_sizes, int n_in,
                              void* d_out, int out_size, void* d_ws, size_t ws_size,
                              hipStream_t stream)
{
    (void)in_sizes; (void)n_in; (void)out_size; (void)ws_size;
    const int*   x    = (const int*)  d_in[0];
    const float* mv   = (const float*)d_in[1];
    const float* Wihf = (const float*)d_in[2];
    const float* Whhf = (const float*)d_in[3];
    const float* bihf = (const float*)d_in[4];
    const float* bhhf = (const float*)d_in[5];
    const float* Wihb = (const float*)d_in[6];
    const float* Whhb = (const float*)d_in[7];
    const float* bihb = (const float*)d_in[8];
    const float* bhhb = (const float*)d_in[9];
    const float* fcw  = (const float*)d_in[10];
    const float* fcb  = (const float*)d_in[11];
    const float* fc2w = (const float*)d_in[12];
    const float* fc2b = (const float*)d_in[13];
    float* out = (float*)d_out;

    char* ws = (char*)d_ws; size_t off = 0;
    auto alloc = [&](size_t bytes) -> void* {
        void* p = ws + off; off = (off + bytes + 255) & ~(size_t)255; return p;
    };

    _Float16* mvn16   = (_Float16*)alloc((size_t)CP * VEC * 2);
    _Float16* mv16    = (_Float16*)alloc((size_t)CP * VEC * 2);
    _Float16* wihf16  = (_Float16*)alloc((size_t)HID * VEC * 2);
    _Float16* wihb16  = (_Float16*)alloc((size_t)HID * VEC * 2);
    _Float16* whhf16  = (_Float16*)alloc((size_t)HID * HID * 2);
    _Float16* whhb16  = (_Float16*)alloc((size_t)HID * HID * 2);
    _Float16* fcw16   = (_Float16*)alloc((size_t)CP * (2 * HID) * 2);
    _Float16* fc2w16  = (_Float16*)alloc((size_t)CP * CP * 2);
    float*    biasf   = (float*)   alloc(HID * 4);
    float*    biasb   = (float*)   alloc(HID * 4);
    int*      idxf    = (int*)     alloc((size_t)TSTEPS * BATCH * 4);
    int*      idxb    = (int*)     alloc((size_t)TSTEPS * BATCH * 4);
    float*    att     = (float*)   alloc((size_t)BATCH * CP * 4);
    float*    xpf     = (float*)   alloc((size_t)TSTEPS * BATCH * HID * 4);
    float*    xpb     = (float*)   alloc((size_t)TSTEPS * BATCH * HID * 4);
    _Float16* hfA     = (_Float16*)alloc((size_t)BATCH * HID * 2);
    _Float16* hfB     = (_Float16*)alloc((size_t)BATCH * HID * 2);
    _Float16* hbA     = (_Float16*)alloc((size_t)BATCH * HID * 2);
    _Float16* hbB     = (_Float16*)alloc((size_t)BATCH * HID * 2);
    _Float16* out4    = (_Float16*)alloc((size_t)BATCH * 2 * HID * 2);
    _Float16* out1    = (_Float16*)alloc((size_t)BATCH * CP * 2);

    auto gemm = [&](const _Float16* A, int lda, const int* gi,
                    const _Float16* W, int ldb, int M, int N, int NP, int K,
                    const float* bias, const float* resid, int ldr,
                    const float* mulp, int ldm, int act,
                    float* oF, int ldoF, _Float16* oH, int ldoH) {
        dim3 g(M / 128, NP / 128), b(256);
        gemm_f16_wmma<<<g, b, 0, stream>>>(A, lda, gi, W, ldb, N, K, bias,
                                           resid, ldr, mulp, ldm, act,
                                           oF, ldoF, oH, ldoH);
    };

    // --- precompute f16 operands ---
    normalize_kernel<<<CP, 256, 0, stream>>>(mv, mvn16, mv16);
    auto cvt = [&](const float* s, int R, int C, _Float16* d, int RP, int CS) {
        size_t tot = (size_t)RP * CS;
        cvt_pad_kernel<<<(int)((tot + 255) / 256), 256, 0, stream>>>(s, R, C, d, RP, CS);
    };
    cvt(Wihf, HID, VEC, wihf16, HID, VEC);
    cvt(Wihb, HID, VEC, wihb16, HID, VEC);
    cvt(Whhf, HID, HID, whhf16, HID, HID);
    cvt(Whhb, HID, HID, whhb16, HID, HID);
    cvt(fcw,  CLASSN, 2 * HID, fcw16,  CP, 2 * HID);
    cvt(fc2w, CLASSN, CLASSN,  fc2w16, CP, CP);
    build_idx_kernel<<<(TSTEPS * BATCH + 255) / 256, 256, 0, stream>>>(x, idxf, idxb);
    bias_kernel<<<2, 256, 0, stream>>>(bihf, bhhf, bihb, bhhb, biasf, biasb);
    {
        size_t n = (size_t)BATCH * HID;
        zero_h_kernel<<<(int)((n + 255) / 256), 256, 0, stream>>>(hfA, n);
        zero_h_kernel<<<(int)((n + 255) / 256), 256, 0, stream>>>(hbA, n);
    }

    // --- att = sigmoid(mv_n[x[:,4]] @ mv_n^T) ; idxb slot 0 == x[:,4] ---
    gemm(mvn16, VEC, idxb, mvn16, VEC, BATCH, CLASSN, CP, VEC,
         nullptr, nullptr, 0, nullptr, 0, 2, att, CP, nullptr, 0);

    // --- xp = emb @ W_ih^T + (b_ih+b_hh), gathered rows, 5 steps at once ---
    gemm(mv16, VEC, idxf, wihf16, VEC, TSTEPS * BATCH, HID, HID, VEC,
         biasf, nullptr, 0, nullptr, 0, 0, xpf, HID, nullptr, 0);
    gemm(mv16, VEC, idxb, wihb16, VEC, TSTEPS * BATCH, HID, HID, VEC,
         biasb, nullptr, 0, nullptr, 0, 0, xpb, HID, nullptr, 0);

    // --- forward RNN: t = 0..4 ; h = tanh(xp_t + h @ W_hh^T) (ping-pong) ---
    {
        _Float16* hin = hfA; _Float16* hout = hfB;
        for (int t = 0; t < TSTEPS; ++t) {
            gemm(hin, HID, nullptr, whhf16, HID, BATCH, HID, HID, HID,
                 nullptr, xpf + (size_t)t * BATCH * HID, HID,
                 nullptr, 0, 1, nullptr, 0, hout, HID);
            _Float16* tmp = hin; hin = hout; hout = tmp;
        }
        // final (hs_f[4]) is in hfB (5 steps -> last write was hfB)
    }
    // --- backward RNN: slots s = 4..0 (t = 8..4) ---
    {
        _Float16* hin = hbA; _Float16* hout = hbB;
        for (int s = TSTEPS - 1; s >= 0; --s) {
            gemm(hin, HID, nullptr, whhb16, HID, BATCH, HID, HID, HID,
                 nullptr, xpb + (size_t)s * BATCH * HID, HID,
                 nullptr, 0, 1, nullptr, 0, hout, HID);
            _Float16* tmp = hin; hin = hout; hout = tmp;
        }
    }

    pack_kernel<<<(BATCH * 2 * HID + 255) / 256, 256, 0, stream>>>(hfB, hbB, out4);

    // --- out1 = (out4 @ fc_w^T + fc_b) * att  (f16, K-padding zeroed) ---
    gemm(out4, 2 * HID, nullptr, fcw16, 2 * HID, BATCH, CLASSN, CP, 2 * HID,
         fcb, nullptr, 0, att, CP, 0, nullptr, 0, out1, CP);

    // --- out = out1 @ fc2_w^T + fc2_b ---
    gemm(out1, CP, nullptr, fc2w16, CP, BATCH, CLASSN, CP, CP,
         fc2b, nullptr, 0, nullptr, 0, 0, out, CLASSN, nullptr, 0);
}
